// LayerNet_59012850647115
// MI455X (gfx1250) — compile-verified, matrix operand order checked by reading, not run on hardware
//
#include <hip/hip_runtime.h>
#include <hip/hip_bf16.h>

typedef __attribute__((ext_vector_type(16))) _Float16 v16h;
typedef __attribute__((ext_vector_type(8)))  _Float16 v8h;
typedef __attribute__((ext_vector_type(8)))  float    v8f;

#define HW_FULL 65536
#define W_FULL  256
#define NSPP    8

// ---------------------------------------------------------------------------
// Weight packer: f32 OIHW -> f16 WMMA A-fragment order
// packed index = ((((tap*KC + kc)*MT + mtile)*32 + lane)*16 + h)
// lane<16 -> K in {o..o+7, 16+o..16+o+7} with o=0 ; lane>=16 -> o=8
// ---------------------------------------------------------------------------
__global__ void __launch_bounds__(256) pack_weights(
    const float* __restrict__ src, _Float16* __restrict__ dst,
    int Cout, int Cin, int khw, int total)
{
  int t = blockIdx.x * 256 + threadIdx.x;
  if (t >= total) return;
  int h    = t & 15;
  int lane = (t >> 4) & 31;
  int r    = t >> 9;
  int MT = Cout >> 4;
  int KC = Cin >> 5;
  int mtile = r % MT; r /= MT;
  int kc    = r % KC;
  int tap   = r / KC;
  int o  = (lane >> 4) * 8;
  int Kk = (h < 8) ? (o + h) : (16 + o + (h - 8));
  int cin  = (kc << 5) + Kk;
  int cout = (mtile << 4) + (lane & 15);
  dst[t] = (_Float16)src[((size_t)cout * Cin + cin) * khw + tap];
}

// ---------------------------------------------------------------------------
// WMMA conv, register-tiled 2x2: each wave computes 32 couts x 32 pixels
// (4 accumulators), sharing A fragments across 2 B fragments and vice versa.
// NTAPS = 9 (3x3 SAME, fully unrolled taps) or 1 (1x1 GEMM).
// Input = concat(inA[Ca], inB[Cb]) NHWC f16; inB pixel index masked by bmask
// (broadcast of ctx across samples). All spatial dims are powers of two.
// Block: 256 threads = 8 waves -> 32 couts x 256 pixels per block.
// ---------------------------------------------------------------------------
template <int NTAPS>
__global__ void __launch_bounds__(256) conv_wmma(
    const _Float16* __restrict__ inA, int Ca,
    const _Float16* __restrict__ inB, int Cb,
    int H, int wshift, int N, int bmask,
    const _Float16* __restrict__ wpk,
    const float* __restrict__ bias,
    _Float16* __restrict__ out, int Cout,
    int relu)
{
  const int lane = threadIdx.x & 31;
  const int wv   = threadIdx.x >> 5;
  const int mt0  = blockIdx.y * 2;              // two consecutive 16-cout tiles
  const int p0   = blockIdx.x * 256 + wv * 32;  // this wave's 32 pixels
  const int n    = lane & 15;
  const int o    = (lane >> 4) * 8;
  const int W    = 1 << wshift;

  int  p_[2], pv_[2], y_[2], x_[2];
  bool val_[2];
  #pragma unroll
  for (int t = 0; t < 2; ++t) {
    p_[t]   = p0 + t * 16 + n;
    val_[t] = (p_[t] < N);
    pv_[t]  = val_[t] ? p_[t] : 0;
    if (NTAPS == 9) { y_[t] = pv_[t] >> wshift; x_[t] = pv_[t] & (W - 1); }
    else            { y_[t] = 0;                x_[t] = 0;                }
  }

  const int Cin = Ca + Cb;
  const int KC  = Cin >> 5;
  const int MT  = Cout >> 4;

  v8f acc[2][2] = {};

  for (int kc = 0; kc < KC; ++kc) {
    const int cbase = kc << 5;
    const _Float16* src; int srcC, coff; bool bsel;
    if (cbase < Ca) { src = inA; srcC = Ca; coff = cbase;      bsel = false; }
    else            { src = inB; srcC = Cb; coff = cbase - Ca; bsel = true;  }

    #pragma unroll
    for (int tap = 0; tap < NTAPS; ++tap) {
      const _Float16* ap =
          wpk + (((size_t)tap * KC + kc) * MT + mt0) * 512 + lane * 16;
      v16h a0 = *(const v16h*)ap;
      v16h a1 = *(const v16h*)(ap + 512);
      if (tap == 0 && kc + 1 < KC)
        __builtin_prefetch(wpk + (((size_t)(kc + 1)) * MT + mt0) * 512, 0, 0);

      v16h b_[2];
      #pragma unroll
      for (int t = 0; t < 2; ++t) {
        int q; bool ok;
        if (NTAPS == 9) {
          const int dy = tap / 3 - 1, dx = tap % 3 - 1;   // compile-time
          int yy = y_[t] + dy, xx = x_[t] + dx;
          ok = val_[t] && yy >= 0 && yy < H && xx >= 0 && xx < W;
          q  = ok ? ((yy << wshift) + xx) : 0;
        } else {
          q = pv_[t]; ok = val_[t];
        }
        if (bsel) q &= bmask;
        if (ok) {
          const _Float16* bp = src + (size_t)q * srcC + coff + o;
          v8h lo = *(const v8h*)bp;
          v8h hh = *(const v8h*)(bp + 16);
          #pragma unroll
          for (int i = 0; i < 8; ++i) { b_[t][i] = lo[i]; b_[t][8 + i] = hh[i]; }
        } else {
          #pragma unroll
          for (int i = 0; i < 16; ++i) b_[t][i] = (_Float16)0.f;
        }
      }
      acc[0][0] = __builtin_amdgcn_wmma_f32_16x16x32_f16(
          false, a0, false, b_[0], (short)0, acc[0][0], false, false);
      acc[0][1] = __builtin_amdgcn_wmma_f32_16x16x32_f16(
          false, a0, false, b_[1], (short)0, acc[0][1], false, false);
      acc[1][0] = __builtin_amdgcn_wmma_f32_16x16x32_f16(
          false, a1, false, b_[0], (short)0, acc[1][0], false, false);
      acc[1][1] = __builtin_amdgcn_wmma_f32_16x16x32_f16(
          false, a1, false, b_[1], (short)0, acc[1][1], false, false);
    }
  }

  #pragma unroll
  for (int mt = 0; mt < 2; ++mt) {
    const int cb = (mt0 + mt) * 16 + o;
    float bv[8];
    #pragma unroll
    for (int r = 0; r < 8; ++r) bv[r] = bias[cb + r];
    #pragma unroll
    for (int pt = 0; pt < 2; ++pt) {
      if (!val_[pt]) continue;
      v8h ov;
      #pragma unroll
      for (int r = 0; r < 8; ++r) {
        float v = acc[mt][pt][r] + bv[r];
        if (relu) v = fmaxf(v, 0.f);
        ov[r] = (_Float16)v;
      }
      *(v8h*)(out + (size_t)p_[pt] * Cout + cb) = ov;
    }
  }
}

// ---------------------------------------------------------------------------
// Input assembly + red1 MLP (18->32->32->32, all ReLU), per-sample emb (f16
// NHWC per sample) and mean embedding (UNet input).
// ---------------------------------------------------------------------------
__global__ void __launch_bounds__(256) embed_kernel(
    const float* __restrict__ color, const float* __restrict__ nd,
    const float* __restrict__ mv, const float* __restrict__ alb,
    const float* __restrict__ spec, const float* __restrict__ uvt,
    const float* __restrict__ W0, const float* __restrict__ B0,
    const float* __restrict__ W1, const float* __restrict__ B1,
    const float* __restrict__ W2, const float* __restrict__ B2,
    _Float16* __restrict__ emb, _Float16* __restrict__ meanemb)
{
  __shared__ float sW0[32 * 18]; __shared__ float sB0[32];
  __shared__ float sW1[32 * 32]; __shared__ float sB1[32];
  __shared__ float sW2[32 * 32]; __shared__ float sB2[32];
  for (int i = threadIdx.x; i < 32 * 18; i += 256) sW0[i] = W0[i];
  for (int i = threadIdx.x; i < 32 * 32; i += 256) { sW1[i] = W1[i]; sW2[i] = W2[i]; }
  if (threadIdx.x < 32) {
    sB0[threadIdx.x] = B0[threadIdx.x];
    sB1[threadIdx.x] = B1[threadIdx.x];
    sB2[threadIdx.x] = B2[threadIdx.x];
  }
  __syncthreads();

  int p = blockIdx.x * 256 + threadIdx.x;
  if (p >= HW_FULL) return;
  const int HW = HW_FULL;

  float msum[32] = {};
  for (int s = 0; s < NSPP; ++s) {
    float in[18];
    {
      size_t b = (size_t)s * 3 * HW + p;
      in[0] = log1pf(fmaxf(color[b], 0.f));
      in[1] = log1pf(fmaxf(color[b + HW], 0.f));
      in[2] = log1pf(fmaxf(color[b + 2 * (size_t)HW], 0.f));
    }
    { size_t b = (size_t)s * 4 * HW + p;
      for (int k = 0; k < 4; ++k) in[3 + k] = nd[b + (size_t)k * HW]; }
    { size_t b = (size_t)s * 2 * HW + p;
      for (int k = 0; k < 2; ++k) in[7 + k] = mv[b + (size_t)k * HW]; }
    { size_t b = (size_t)s * 3 * HW + p;
      for (int k = 0; k < 3; ++k) in[9 + k]  = alb[b + (size_t)k * HW];
      for (int k = 0; k < 3; ++k) in[12 + k] = spec[b + (size_t)k * HW];
      for (int k = 0; k < 3; ++k) in[15 + k] = uvt[b + (size_t)k * HW]; }

    float h[32], g[32];
    for (int oc = 0; oc < 32; ++oc) {
      float a = sB0[oc];
      for (int i = 0; i < 18; ++i) a += sW0[oc * 18 + i] * in[i];
      h[oc] = fmaxf(a, 0.f);
    }
    for (int oc = 0; oc < 32; ++oc) {
      float a = sB1[oc];
      for (int i = 0; i < 32; ++i) a += sW1[oc * 32 + i] * h[i];
      g[oc] = fmaxf(a, 0.f);
    }
    for (int oc = 0; oc < 32; ++oc) {
      float a = sB2[oc];
      for (int i = 0; i < 32; ++i) a += sW2[oc * 32 + i] * g[i];
      h[oc] = fmaxf(a, 0.f);
    }
    _Float16* ep = emb + ((size_t)s * HW + p) * 32;
    for (int c = 0; c < 32; ++c) { ep[c] = (_Float16)h[c]; msum[c] += h[c]; }
  }
  _Float16* mp = meanemb + (size_t)p * 32;
  for (int c = 0; c < 32; ++c) mp[c] = (_Float16)(msum[c] * 0.125f);
}

// ---------------------------------------------------------------------------
// 2x2 max pool / nearest 2x upsample (NHWC f16)
// ---------------------------------------------------------------------------
__global__ void __launch_bounds__(256) maxpool2(
    const _Float16* __restrict__ in, _Float16* __restrict__ out,
    int H2, int W2, int C)
{
  int t = blockIdx.x * 256 + threadIdx.x;
  int total = H2 * W2 * C;
  if (t >= total) return;
  int c = t % C; int xy = t / C;
  int x = xy % W2, y = xy / W2;
  int W = W2 * 2;
  const _Float16* pp = in + ((size_t)(2 * y) * W + 2 * x) * C + c;
  float a = (float)pp[0], b = (float)pp[C];
  float d = (float)pp[(size_t)W * C], e = (float)pp[(size_t)W * C + C];
  out[t] = (_Float16)fmaxf(fmaxf(a, b), fmaxf(d, e));
}

__global__ void __launch_bounds__(256) upsample2(
    const _Float16* __restrict__ in, _Float16* __restrict__ out,
    int Ho, int Wo, int C)
{
  int t = blockIdx.x * 256 + threadIdx.x;
  int total = Ho * Wo * C;
  if (t >= total) return;
  int c = t % C; int xy = t / C;
  int x = xy % Wo, y = xy / Wo;
  out[t] = in[((size_t)(y >> 1) * (Wo >> 1) + (x >> 1)) * C + c];
}

// ---------------------------------------------------------------------------
// part tail: 32 -> 16 (relu) -> 2, softmax over the 2, /nspp
// ---------------------------------------------------------------------------
__global__ void __launch_bounds__(256) part_tail(
    const _Float16* __restrict__ p1,
    const float* __restrict__ W1, const float* __restrict__ B1,
    const float* __restrict__ W2, const float* __restrict__ B2,
    float* __restrict__ wgt, int total)
{
  __shared__ float sW1[16 * 32]; __shared__ float sB1[16];
  __shared__ float sW2[2 * 16];  __shared__ float sB2[2];
  for (int i = threadIdx.x; i < 16 * 32; i += 256) sW1[i] = W1[i];
  if (threadIdx.x < 32) sW2[threadIdx.x] = W2[threadIdx.x];
  if (threadIdx.x < 16) sB1[threadIdx.x] = B1[threadIdx.x];
  if (threadIdx.x < 2)  sB2[threadIdx.x] = B2[threadIdx.x];
  __syncthreads();

  int t = blockIdx.x * 256 + threadIdx.x;
  if (t >= total) return;
  const _Float16* v = p1 + (size_t)t * 32;
  float h[16];
  for (int oc = 0; oc < 16; ++oc) {
    float a = sB1[oc];
    for (int i = 0; i < 32; ++i) a += sW1[oc * 32 + i] * (float)v[i];
    h[oc] = fmaxf(a, 0.f);
  }
  float l0 = sB2[0], l1 = sB2[1];
  for (int i = 0; i < 16; ++i) { l0 += sW2[i] * h[i]; l1 += sW2[16 + i] * h[i]; }
  float m = fmaxf(l0, l1);
  float e0 = expf(l0 - m), e1 = expf(l1 - m);
  float inv = 1.f / (e0 + e1);
  wgt[(size_t)t * 2]     = e0 * inv * 0.125f;
  wgt[(size_t)t * 2 + 1] = e1 * inv * 0.125f;
}

// ---------------------------------------------------------------------------
// Per-pixel reductions over samples: l_rad, l_e, l_w, l_n -> stacked + le
// ---------------------------------------------------------------------------
__global__ void __launch_bounds__(256) aggregate_kernel(
    const float* __restrict__ color, const _Float16* __restrict__ emb,
    const float* __restrict__ wgt,
    _Float16* __restrict__ lebuf, float* __restrict__ stacked)
{
  int p = blockIdx.x * 256 + threadIdx.x;
  if (p >= HW_FULL) return;
  const int HW = HW_FULL;

  float w0[NSPP], w1[NSPP];
  float lw0 = 0.f, lw1 = 0.f;
  for (int s = 0; s < NSPP; ++s) {
    w0[s] = wgt[((size_t)s * HW + p) * 2];
    w1[s] = wgt[((size_t)s * HW + p) * 2 + 1];
    lw0 += w0[s]; lw1 += w1[s];
  }
  float lr0[3] = {}, lr1[3] = {};
  for (int s = 0; s < NSPP; ++s)
    for (int c = 0; c < 3; ++c) {
      float cv = color[((size_t)s * 3 + c) * HW + p];
      lr0[c] += cv * w0[s]; lr1[c] += cv * w1[s];
    }
  float le0[32] = {}, le1[32] = {};
  for (int s = 0; s < NSPP; ++s) {
    const _Float16* ep = emb + ((size_t)s * HW + p) * 32;
    for (int c = 0; c < 32; ++c) {
      float e = (float)ep[c];
      le0[c] += e * w0[s]; le1[c] += e * w1[s];
    }
  }
  _Float16* l0 = lebuf + (size_t)p * 32;
  _Float16* l1 = lebuf + (size_t)HW * 32 + (size_t)p * 32;
  for (int c = 0; c < 32; ++c) { l0[c] = (_Float16)le0[c]; l1[c] = (_Float16)le1[c]; }

  float* s0 = stacked + (size_t)p * 5;
  float* s1 = stacked + (size_t)HW * 5 + (size_t)p * 5;
  s0[0] = lr0[0]; s0[1] = lr0[1]; s0[2] = lr0[2]; s0[3] = lw0; s0[4] = lw0 + lw1;
  s1[0] = lr1[0]; s1[1] = lr1[1]; s1[2] = lr1[2]; s1[3] = lw1; s1[4] = lw1;
}

// ---------------------------------------------------------------------------
// kgen tail: 128 -> 25 (no act)
// ---------------------------------------------------------------------------
__global__ void __launch_bounds__(256) kgen_tail(
    const _Float16* __restrict__ feat,
    const float* __restrict__ Wt, const float* __restrict__ Bt,
    float* __restrict__ kern, int total)
{
  __shared__ float sW[25 * 128]; __shared__ float sB[25];
  for (int i = threadIdx.x; i < 25 * 128; i += 256) sW[i] = Wt[i];
  if (threadIdx.x < 25) sB[threadIdx.x] = Bt[threadIdx.x];
  __syncthreads();

  int t = blockIdx.x * 256 + threadIdx.x;
  if (t >= total) return;
  const _Float16* v = feat + (size_t)t * 128;
  for (int u = 0; u < 25; ++u) {
    float a = sB[u];
    for (int i = 0; i < 128; ++i) a += sW[u * 128 + i] * (float)v[i];
    kern[(size_t)t * 25 + u] = a;
  }
}

// kern = exp(kern - max over all 50 channels), in place
__global__ void __launch_bounds__(256) kern_norm(float* __restrict__ kern)
{
  int p = blockIdx.x * 256 + threadIdx.x;
  if (p >= HW_FULL) return;
  const size_t joff = (size_t)HW_FULL * 25;
  float m = -3.4e38f;
  for (int u = 0; u < 25; ++u) {
    m = fmaxf(m, kern[(size_t)p * 25 + u]);
    m = fmaxf(m, kern[joff + (size_t)p * 25 + u]);
  }
  for (int u = 0; u < 25; ++u) {
    kern[(size_t)p * 25 + u]        = expf(kern[(size_t)p * 25 + u] - m);
    kern[joff + (size_t)p * 25 + u] = expf(kern[joff + (size_t)p * 25 + u] - m);
  }
}

// ---------------------------------------------------------------------------
// 5x5 splat (expressed as gather) + front-to-back composite -> NCHW f32 out
// ---------------------------------------------------------------------------
__global__ void __launch_bounds__(256) splat_composite(
    const float* __restrict__ stacked, const float* __restrict__ kern,
    float* __restrict__ out)
{
  int p = blockIdx.x * 256 + threadIdx.x;
  if (p >= HW_FULL) return;
  const int HW = HW_FULL, W = W_FULL;
  int y = p >> 8, x = p & 255;

  float col0 = 0.f, col1 = 0.f, col2 = 0.f, kacc = 1.f;
  for (int j = 0; j < 2; ++j) {
    const float* bs = stacked + (size_t)j * HW * 5;
    const float* bk = kern    + (size_t)j * HW * 25;
    float a0 = 0.f, a1 = 0.f, a2 = 0.f, a3 = 0.f, a4 = 0.f;
    for (int dy = -2; dy <= 2; ++dy) {
      int sy = y - dy;
      if (sy < 0 || sy >= W) continue;
      for (int dx = -2; dx <= 2; ++dx) {
        int sx = x - dx;
        if (sx < 0 || sx >= W) continue;
        int sp = (sy << 8) + sx;
        int idx = (dy + 2) * 5 + (dx + 2);
        float kv = bk[(size_t)sp * 25 + idx];
        const float* st = bs + (size_t)sp * 5;
        a0 += st[0] * kv; a1 += st[1] * kv; a2 += st[2] * kv;
        a3 += st[3] * kv; a4 += st[4] * kv;
      }
    }
    float inv = 1.f / (a4 + 1e-6f);
    float alpha = a3 * inv;
    col0 += a0 * inv * kacc;
    col1 += a1 * inv * kacc;
    col2 += a2 * inv * kacc;
    kacc *= (1.f - alpha);
  }
  out[p]          = col0;
  out[HW + p]     = col1;
  out[2 * HW + p] = col2;
}

// ---------------------------------------------------------------------------
// Host orchestration
// ---------------------------------------------------------------------------
struct LayerPack { int widx; int Cout; int Cin; int khw; };
static const LayerPack g_packs[21] = {
  {24, 64,   32, 9}, {26, 64,   64, 9}, {28, 128,  64, 9}, {30, 256, 128, 9},
  {32, 512, 256, 9}, {34, 512, 512, 9}, {36, 512, 512, 9},
  {38, 512,1024, 9}, {40, 512, 512, 9}, {42, 256,1024, 9}, {44, 256, 256, 9},
  {46, 128, 512, 9}, {48, 128, 128, 9}, {50, 128, 256, 9}, {52, 128, 128, 9},
  {54, 128, 192, 9}, {56, 128, 128, 9}, {58, 128, 128, 9},
  {12, 32,  160, 1}, {18, 128, 160, 1}, {20, 128, 128, 1},
};

extern "C" void kernel_launch(void* const* d_in, const int* in_sizes, int n_in,
                              void* d_out, int out_size, void* d_ws, size_t ws_size,
                              hipStream_t stream) {
  (void)in_sizes; (void)n_in; (void)out_size; (void)ws_size;
  const int HW = HW_FULL;
  char* ws = (char*)d_ws;
  size_t off = 0;
  auto alloc = [&](size_t bytes) -> char* {
    char* r = ws + off;
    off = (off + bytes + 255) & ~(size_t)255;
    return r;
  };

  // packed weights
  _Float16* wp[21];
  for (int i = 0; i < 21; ++i)
    wp[i] = (_Float16*)alloc(2ull * g_packs[i].Cout * g_packs[i].Cin * g_packs[i].khw);

  // activations
  _Float16* emb     = (_Float16*)alloc(2ull * NSPP * HW * 32);
  _Float16* meanemb = (_Float16*)alloc(2ull * HW * 32);
  _Float16* skip0   = (_Float16*)alloc(2ull * HW * 64);
  _Float16* skip1   = (_Float16*)alloc(2ull * 16384 * 128);
  _Float16* skip2   = (_Float16*)alloc(2ull * 4096 * 256);
  _Float16* skip3   = (_Float16*)alloc(2ull * 1024 * 512);
  _Float16* skip4   = (_Float16*)alloc(2ull * 256 * 512);
  _Float16* bufA    = (_Float16*)alloc(2ull * HW * 128);
  _Float16* bufB    = (_Float16*)alloc(2ull * HW * 128);
  _Float16* bufU    = (_Float16*)alloc(2ull * HW * 128);
  _Float16* ctx     = (_Float16*)alloc(2ull * HW * 128);
  _Float16* part1   = (_Float16*)alloc(2ull * NSPP * HW * 32);
  float*    wgt     = (float*)   alloc(4ull * NSPP * HW * 2);
  _Float16* lebuf   = (_Float16*)alloc(2ull * 2 * HW * 32);
  float*    stacked = (float*)   alloc(4ull * 2 * HW * 5);
  _Float16* kA      = (_Float16*)alloc(2ull * HW * 128);
  _Float16* kB      = (_Float16*)alloc(2ull * 2 * HW * 128);
  float*    kern    = (float*)   alloc(4ull * 2 * HW * 25);

  // pack all WMMA weights (f32 -> fragment-ordered f16)
  for (int i = 0; i < 21; ++i) {
    int total = g_packs[i].Cout * g_packs[i].Cin * g_packs[i].khw;
    pack_weights<<<(total + 255) / 256, 256, 0, stream>>>(
        (const float*)d_in[g_packs[i].widx], wp[i],
        g_packs[i].Cout, g_packs[i].Cin, g_packs[i].khw, total);
  }

  // input assembly + red1 MLP
  embed_kernel<<<HW / 256, 256, 0, stream>>>(
      (const float*)d_in[0], (const float*)d_in[1], (const float*)d_in[2],
      (const float*)d_in[3], (const float*)d_in[4], (const float*)d_in[5],
      (const float*)d_in[6], (const float*)d_in[7], (const float*)d_in[8],
      (const float*)d_in[9], (const float*)d_in[10], (const float*)d_in[11],
      emb, meanemb);

  auto conv3 = [&](const _Float16* inA, int Ca, const _Float16* inB, int Cb,
                   int Hh, int wshift, int N, int wpi, int bidx,
                   _Float16* out, int Cout, int relu) {
    dim3 g((N + 255) / 256, Cout / 32);
    conv_wmma<9><<<g, 256, 0, stream>>>(inA, Ca, inB, Cb, Hh, wshift, N, N - 1,
                                        wp[wpi], (const float*)d_in[bidx],
                                        out, Cout, relu);
  };
  auto gemm = [&](const _Float16* inA, int Ca, const _Float16* inB, int Cb,
                  int N, int bmask, int wpi, int bidx,
                  _Float16* out, int Cout, int relu) {
    dim3 g((N + 255) / 256, Cout / 32);
    conv_wmma<1><<<g, 256, 0, stream>>>(inA, Ca, inB, Cb, 1, 0, N, bmask,
                                        wp[wpi], (const float*)d_in[bidx],
                                        out, Cout, relu);
  };
  auto pool = [&](const _Float16* in, _Float16* out, int H2, int W2, int C) {
    int total = H2 * W2 * C;
    maxpool2<<<(total + 255) / 256, 256, 0, stream>>>(in, out, H2, W2, C);
  };
  auto up = [&](const _Float16* in, _Float16* out, int Ho, int Wo, int C) {
    int total = Ho * Wo * C;
    upsample2<<<(total + 255) / 256, 256, 0, stream>>>(in, out, Ho, Wo, C);
  };

  // ---------------- UNet encoder ----------------
  conv3(meanemb, 32, meanemb, 0, 256, 8, 65536, 0, 25, bufA, 64, 1);
  conv3(bufA, 64, bufA, 0, 256, 8, 65536, 1, 27, skip0, 64, 1);
  pool(skip0, bufA, 128, 128, 64);
  conv3(bufA, 64, bufA, 0, 128, 7, 16384, 2, 29, skip1, 128, 1);
  pool(skip1, bufA, 64, 64, 128);
  conv3(bufA, 128, bufA, 0, 64, 6, 4096, 3, 31, skip2, 256, 1);
  pool(skip2, bufA, 32, 32, 256);
  conv3(bufA, 256, bufA, 0, 32, 5, 1024, 4, 33, skip3, 512, 1);
  pool(skip3, bufA, 16, 16, 512);
  conv3(bufA, 512, bufA, 0, 16, 4, 256, 5, 35, skip4, 512, 1);
  pool(skip4, bufA, 8, 8, 512);
  // bottleneck
  conv3(bufA, 512, bufA, 0, 8, 3, 64, 6, 37, bufB, 512, 1);

  // ---------------- UNet decoder ----------------
  up(bufB, bufU, 16, 16, 512);
  conv3(bufU, 512, skip4, 512, 16, 4, 256, 7, 39, bufA, 512, 1);
  conv3(bufA, 512, bufA, 0, 16, 4, 256, 8, 41, bufB, 512, 1);

  up(bufB, bufU, 32, 32, 512);
  conv3(bufU, 512, skip3, 512, 32, 5, 1024, 9, 43, bufA, 256, 1);
  conv3(bufA, 256, bufA, 0, 32, 5, 1024, 10, 45, bufB, 256, 1);

  up(bufB, bufU, 64, 64, 256);
  conv3(bufU, 256, skip2, 256, 64, 6, 4096, 11, 47, bufA, 128, 1);
  conv3(bufA, 128, bufA, 0, 64, 6, 4096, 12, 49, bufB, 128, 1);

  up(bufB, bufU, 128, 128, 128);
  conv3(bufU, 128, skip1, 128, 128, 7, 16384, 13, 51, bufA, 128, 1);
  conv3(bufA, 128, bufA, 0, 128, 7, 16384, 14, 53, bufB, 128, 1);

  up(bufB, bufU, 256, 256, 128);
  conv3(bufU, 128, skip0, 64, 256, 8, 65536, 15, 55, bufA, 128, 1);
  conv3(bufA, 128, bufA, 0, 256, 8, 65536, 16, 57, bufB, 128, 1);
  // final (no relu)
  conv3(bufB, 128, bufB, 0, 256, 8, 65536, 17, 59, ctx, 128, 0);

  // ---------------- partitioner ----------------
  gemm(emb, 32, ctx, 128, NSPP * HW, HW - 1, 18, 13, part1, 32, 1);
  part_tail<<<(NSPP * HW + 255) / 256, 256, 0, stream>>>(
      part1, (const float*)d_in[14], (const float*)d_in[15],
      (const float*)d_in[16], (const float*)d_in[17], wgt, NSPP * HW);

  aggregate_kernel<<<HW / 256, 256, 0, stream>>>(
      (const float*)d_in[0], emb, wgt, lebuf, stacked);

  // ---------------- kernel generator ----------------
  for (int j = 0; j < 2; ++j) {
    gemm(lebuf + (size_t)j * HW * 32, 32, ctx, 128, HW, HW - 1, 19, 19, kA, 128, 1);
    gemm(kA, 128, kA, 0, HW, HW - 1, 20, 21, kB + (size_t)j * HW * 128, 128, 1);
  }
  kgen_tail<<<(2 * HW + 255) / 256, 256, 0, stream>>>(
      kB, (const float*)d_in[22], (const float*)d_in[23], kern, 2 * HW);
  kern_norm<<<HW / 256, 256, 0, stream>>>(kern);

  // ---------------- splat + composite ----------------
  splat_composite<<<HW / 256, 256, 0, stream>>>(stacked, kern, (float*)d_out);
}
